// GCN_Layer_90907277787237
// MI455X (gfx1250) — matrix operand order repaired and checked by reading, not compiled, not run on hardware
//
#include <hip/hip_runtime.h>

#define B_   32
#define N_   1024
#define DIN  512
#define DMEM 512

typedef __attribute__((ext_vector_type(16))) _Float16 v16h;
typedef __attribute__((ext_vector_type(8)))  _Float16 v8h;
typedef __attribute__((ext_vector_type(8)))  float    v8f;

__device__ __forceinline__ v16h cat8(v8h lo, v8h hi) {
  return __builtin_shufflevector(lo, hi, 0,1,2,3,4,5,6,7,8,9,10,11,12,13,14,15);
}
// A fragment (16-bit A layout): k = (j/8)*16 + hi*8 + (j%8) -> two 16B chunks
__device__ __forceinline__ v16h loadA(const _Float16* __restrict__ row, int kk, int hi) {
  return cat8(*(const v8h*)(row + kk + hi * 8),
              *(const v8h*)(row + kk + 16 + hi * 8));
}
// B fragment (16-bit B layout): k = (j/8)*16 + hi*8 + (j%8) -> two 16B chunks
__device__ __forceinline__ v16h loadB(const _Float16* __restrict__ col, int kk, int hi) {
  return cat8(*(const v8h*)(col + kk + hi * 8),
              *(const v8h*)(col + kk + 16 + hi * 8));
}
#define WMMA(A, Bf, C) __builtin_amdgcn_wmma_f32_16x16x32_f16(false, A, false, Bf, (short)0, C, false, false)

// 8 WMMAs for one 32x64 wave tile against one k-step fragment set
#define DO_WMMA8(AF0, AF1, BF0, BF1, BF2, BF3)          \
  do {                                                  \
    acc00 = WMMA(AF0, BF0, acc00);                      \
    acc10 = WMMA(AF1, BF0, acc10);                      \
    acc01 = WMMA(AF0, BF1, acc01);                      \
    acc11 = WMMA(AF1, BF1, acc11);                      \
    acc02 = WMMA(AF0, BF2, acc02);                      \
    acc12 = WMMA(AF1, BF2, acc12);                      \
    acc03 = WMMA(AF0, BF3, acc03);                      \
    acc13 = WMMA(AF1, BF3, acc13);                      \
  } while (0)

// ---------------------------------------------------------------------------
// adj f32 -> f16 conversion fused with row sums (denom = rowsum + 1)
// ---------------------------------------------------------------------------
__global__ __launch_bounds__(256)
void k_adj_cvt_rowsum(const float* __restrict__ adj, _Float16* __restrict__ adj_h,
                      float* __restrict__ denom, float* __restrict__ rowsum) {
  const int row = blockIdx.x;                       // 0 .. B*N-1
  const float* src = adj + (size_t)row * N_;
  _Float16* dst = adj_h + (size_t)row * N_;
  float s = 0.f;
  for (int i = threadIdx.x; i < N_; i += 256) {
    float v = src[i];
    dst[i] = (_Float16)v;
    s += v;
  }
  __shared__ float red[256];
  red[threadIdx.x] = s;
  __syncthreads();
  for (int w = 128; w > 0; w >>= 1) {
    if (threadIdx.x < w) red[threadIdx.x] += red[threadIdx.x + w];
    __syncthreads();
  }
  if (threadIdx.x == 0) {
    denom[row]  = red[0] + 1.0f;
    rowsum[row] = red[0];
  }
}

// ---------------------------------------------------------------------------
// column sums of adj + mask output: mask = (rowsum + colsum == 0)
// ---------------------------------------------------------------------------
__global__ __launch_bounds__(256)
void k_colsum_mask(const float* __restrict__ adj, const float* __restrict__ rowsum,
                   float* __restrict__ mask_out) {
  const int b = blockIdx.y;
  const int n = blockIdx.x * 256 + threadIdx.x;
  const float* ab = adj + (size_t)b * N_ * N_;
  float s = 0.f;
  for (int m = 0; m < N_; ++m) s += ab[(size_t)m * N_ + n];
  const float tot = s + rowsum[b * N_ + n];
  mask_out[b * N_ + n] = (tot == 0.0f) ? 1.0f : 0.0f;
}

// ---------------------------------------------------------------------------
// generic f32 -> f16 (weights)
// ---------------------------------------------------------------------------
__global__ __launch_bounds__(256)
void k_f32_to_f16(const float* __restrict__ src, _Float16* __restrict__ dst, int n) {
  const int i = blockIdx.x * 256 + threadIdx.x;
  if (i < n) dst[i] = (_Float16)src[i];
}

// ---------------------------------------------------------------------------
// input_emb f32 -> x_h (row major [B*N, D]) AND xT_h (transposed [B, D, N])
// LDS-tiled 64x64 transpose, coalesced on both sides.
// ---------------------------------------------------------------------------
__global__ __launch_bounds__(256)
void k_cvt_x_T(const float* __restrict__ src, _Float16* __restrict__ x_h,
               _Float16* __restrict__ xT_h) {
  const int b  = blockIdx.z;
  const int n0 = blockIdx.y * 64;   // node index tile
  const int d0 = blockIdx.x * 64;   // feature index tile
  __shared__ _Float16 tile[64][72];
  const int tid = threadIdx.x;
  #pragma unroll
  for (int rep = 0; rep < 16; ++rep) {
    const int idx = rep * 256 + tid;
    const int r = idx >> 6;       // n within tile
    const int c = idx & 63;       // d within tile
    const size_t g = ((size_t)(b * N_ + n0 + r)) * DIN + d0 + c;
    const float v = src[g];
    const _Float16 h = (_Float16)v;
    tile[r][c] = h;
    x_h[g] = h;
  }
  __syncthreads();
  #pragma unroll
  for (int rep = 0; rep < 16; ++rep) {
    const int idx = rep * 256 + tid;
    const int rr = idx >> 6;      // d within tile
    const int cc = idx & 63;      // n within tile
    xT_h[((size_t)(b * DIN + d0 + rr)) * N_ + n0 + cc] = tile[cc][rr];
  }
}

// ---------------------------------------------------------------------------
// GEMM1: T[b] = adj[b] (1024x1024) @ x[b] (1024x512) + x[b]
// A fragments from adj (row major), B fragments from xT (feature major);
// per-lane contiguous 16B chunks, no LDS, no barriers.
// wave tile = 32 (m) x 64 (n) = 2x4 WMMA tiles; 8 waves -> 128x128 per WG.
// Two-stage ping-pong in K (no register rotation copies).
// ---------------------------------------------------------------------------
__global__ __launch_bounds__(256)
void k_gemm1(const _Float16* __restrict__ adj_h,
             const _Float16* __restrict__ xT_h,
             const _Float16* __restrict__ x_h,
             _Float16* __restrict__ T_h) {
  const int b    = blockIdx.z;
  const int m0   = blockIdx.y * 128;
  const int n0   = blockIdx.x * 128;
  const int tid  = threadIdx.x;
  const int lane = tid & 31;
  const int wave = tid >> 5;
  const int wm   = wave & 3;     // 4 row-blocks of 32
  const int wn   = wave >> 2;    // 2 col-blocks of 64
  const int lo   = lane & 15;
  const int hi   = lane >> 4;

  const _Float16* A  = adj_h + (size_t)b * N_ * N_;
  const _Float16* XT = xT_h + (size_t)b * DIN * N_;
  const _Float16* X  = x_h  + (size_t)b * N_ * DIN;

  v8f acc00 = {}, acc01 = {}, acc02 = {}, acc03 = {};
  v8f acc10 = {}, acc11 = {}, acc12 = {}, acc13 = {};

  const int rA0 = m0 + wm * 32 + lo;
  const _Float16* arow0 = A + (size_t)rA0 * N_;
  const _Float16* arow1 = arow0 + (size_t)16 * N_;
  const _Float16* bcol0 = XT + (size_t)(n0 + wn * 64 + lo) * N_;
  const _Float16* bcol1 = bcol0 + (size_t)16 * N_;
  const _Float16* bcol2 = bcol0 + (size_t)32 * N_;
  const _Float16* bcol3 = bcol0 + (size_t)48 * N_;

  // stage 0 prologue (k = 0)
  v16h pa0 = loadA(arow0, 0, hi);
  v16h pa1 = loadA(arow1, 0, hi);
  v16h pb0 = loadB(bcol0, 0, hi);
  v16h pb1 = loadB(bcol1, 0, hi);
  v16h pb2 = loadB(bcol2, 0, hi);
  v16h pb3 = loadB(bcol3, 0, hi);

  int kk = 0;
  #pragma unroll 1
  for (; kk < N_ - 64; kk += 64) {
    // stage 1 loads (k+32) -- in flight during stage-0 WMMAs
    v16h qa0 = loadA(arow0, kk + 32, hi);
    v16h qa1 = loadA(arow1, kk + 32, hi);
    v16h qb0 = loadB(bcol0, kk + 32, hi);
    v16h qb1 = loadB(bcol1, kk + 32, hi);
    v16h qb2 = loadB(bcol2, kk + 32, hi);
    v16h qb3 = loadB(bcol3, kk + 32, hi);
    __builtin_prefetch(arow0 + kk + 128, 0, 1);
    __builtin_prefetch(arow1 + kk + 128, 0, 1);

    DO_WMMA8(pa0, pa1, pb0, pb1, pb2, pb3);          // k

    // stage 0 loads (k+64) -- old stage-0 frags are dead, regs recycled
    pa0 = loadA(arow0, kk + 64, hi);
    pa1 = loadA(arow1, kk + 64, hi);
    pb0 = loadB(bcol0, kk + 64, hi);
    pb1 = loadB(bcol1, kk + 64, hi);
    pb2 = loadB(bcol2, kk + 64, hi);
    pb3 = loadB(bcol3, kk + 64, hi);

    DO_WMMA8(qa0, qa1, qb0, qb1, qb2, qb3);          // k + 32
  }
  // tail: kk == N_-64
  {
    v16h qa0 = loadA(arow0, kk + 32, hi);
    v16h qa1 = loadA(arow1, kk + 32, hi);
    v16h qb0 = loadB(bcol0, kk + 32, hi);
    v16h qb1 = loadB(bcol1, kk + 32, hi);
    v16h qb2 = loadB(bcol2, kk + 32, hi);
    v16h qb3 = loadB(bcol3, kk + 32, hi);
    DO_WMMA8(pa0, pa1, pb0, pb1, pb2, pb3);
    DO_WMMA8(qa0, qa1, qb0, qb1, qb2, qb3);
  }

  // epilogue: T = Ax + x (f16 store)
  _Float16* Tb = T_h + (size_t)b * N_ * DIN;
  v8f accs[2][4] = {{acc00, acc01, acc02, acc03}, {acc10, acc11, acc12, acc13}};
  #pragma unroll
  for (int mt = 0; mt < 2; ++mt) {
    const int rbase = m0 + wm * 32 + mt * 16 + hi * 8;
    #pragma unroll
    for (int nt = 0; nt < 4; ++nt) {
      const int c = n0 + wn * 64 + nt * 16 + lo;
      #pragma unroll
      for (int i = 0; i < 8; ++i) {
        const int r = rbase + i;
        Tb[(size_t)r * DIN + c] = (_Float16)(accs[mt][nt][i] + (float)X[(size_t)r * DIN + c]);
      }
    }
  }
}

// ---------------------------------------------------------------------------
// GEMM2: x = relu((T @ W^T + 2b) / denom); writes x (row major) AND xT.
// B fragment column j of W^T = row j of W -> contiguous global loads.
// Same two-stage ping-pong.
// ---------------------------------------------------------------------------
__global__ __launch_bounds__(256)
void k_gemm2(const _Float16* __restrict__ T_h,
             const _Float16* __restrict__ W_h,     // [512 x 512] (out, in)
             const float* __restrict__ Wb,
             const float* __restrict__ denom,
             _Float16* __restrict__ x_out,
             _Float16* __restrict__ xT_out) {
  const int m0   = blockIdx.y * 128;   // over B*N rows
  const int n0   = blockIdx.x * 128;   // over 512 output features
  const int tid  = threadIdx.x;
  const int lane = tid & 31;
  const int wave = tid >> 5;
  const int wm   = wave & 3;
  const int wn   = wave >> 2;
  const int lo   = lane & 15;
  const int hi   = lane >> 4;

  v8f acc00 = {}, acc01 = {}, acc02 = {}, acc03 = {};
  v8f acc10 = {}, acc11 = {}, acc12 = {}, acc13 = {};

  const int rA0 = m0 + wm * 32 + lo;
  const _Float16* arow0 = T_h + (size_t)rA0 * DMEM;
  const _Float16* arow1 = arow0 + (size_t)16 * DMEM;
  const _Float16* brow0 = W_h + (size_t)(n0 + wn * 64 + lo) * DMEM;
  const _Float16* brow1 = brow0 + (size_t)16 * DMEM;
  const _Float16* brow2 = brow0 + (size_t)32 * DMEM;
  const _Float16* brow3 = brow0 + (size_t)48 * DMEM;

  v16h pa0 = loadA(arow0, 0, hi);
  v16h pa1 = loadA(arow1, 0, hi);
  v16h pb0 = loadB(brow0, 0, hi);
  v16h pb1 = loadB(brow1, 0, hi);
  v16h pb2 = loadB(brow2, 0, hi);
  v16h pb3 = loadB(brow3, 0, hi);

  int kk = 0;
  #pragma unroll 1
  for (; kk < DMEM - 64; kk += 64) {
    v16h qa0 = loadA(arow0, kk + 32, hi);
    v16h qa1 = loadA(arow1, kk + 32, hi);
    v16h qb0 = loadB(brow0, kk + 32, hi);
    v16h qb1 = loadB(brow1, kk + 32, hi);
    v16h qb2 = loadB(brow2, kk + 32, hi);
    v16h qb3 = loadB(brow3, kk + 32, hi);

    DO_WMMA8(pa0, pa1, pb0, pb1, pb2, pb3);

    pa0 = loadA(arow0, kk + 64, hi);
    pa1 = loadA(arow1, kk + 64, hi);
    pb0 = loadB(brow0, kk + 64, hi);
    pb1 = loadB(brow1, kk + 64, hi);
    pb2 = loadB(brow2, kk + 64, hi);
    pb3 = loadB(brow3, kk + 64, hi);

    DO_WMMA8(qa0, qa1, qb0, qb1, qb2, qb3);
  }
  {
    v16h qa0 = loadA(arow0, kk + 32, hi);
    v16h qa1 = loadA(arow1, kk + 32, hi);
    v16h qb0 = loadB(brow0, kk + 32, hi);
    v16h qb1 = loadB(brow1, kk + 32, hi);
    v16h qb2 = loadB(brow2, kk + 32, hi);
    v16h qb3 = loadB(brow3, kk + 32, hi);
    DO_WMMA8(pa0, pa1, pb0, pb1, pb2, pb3);
    DO_WMMA8(qa0, qa1, qb0, qb1, qb2, qb3);
  }

  v8f accs[2][4] = {{acc00, acc01, acc02, acc03}, {acc10, acc11, acc12, acc13}};
  #pragma unroll
  for (int mt = 0; mt < 2; ++mt) {
    const int rbase = m0 + wm * 32 + mt * 16 + hi * 8;
    #pragma unroll
    for (int nt = 0; nt < 4; ++nt) {
      const int c = n0 + wn * 64 + nt * 16 + lo;
      const float bias2 = 2.0f * Wb[c];
      #pragma unroll
      for (int i = 0; i < 8; ++i) {
        const int r = rbase + i;                 // flattened row over B*N
        const float v = fmaxf((accs[mt][nt][i] + bias2) / denom[r], 0.f);
        const _Float16 h = (_Float16)v;
        x_out[(size_t)r * DMEM + c] = h;
        const int bb = r >> 10;                  // r / N_
        const int nn = r & (N_ - 1);             // r % N_
        xT_out[((size_t)(bb * DMEM + c)) * N_ + nn] = h;
      }
    }
  }
}

// ---------------------------------------------------------------------------
// LayerNorm over last dim (512), f32 math, f32 output
// ---------------------------------------------------------------------------
__global__ __launch_bounds__(256)
void k_layernorm(const _Float16* __restrict__ x, const float* __restrict__ g,
                 const float* __restrict__ bb, float* __restrict__ out) {
  const int row = blockIdx.x;
  const int tid = threadIdx.x;
  const _Float16* xr = x + (size_t)row * DMEM;
  const float v0 = (float)xr[tid];
  const float v1 = (float)xr[tid + 256];
  __shared__ float rs[256], rq[256];
  rs[tid] = v0 + v1;
  rq[tid] = v0 * v0 + v1 * v1;
  __syncthreads();
  for (int w = 128; w > 0; w >>= 1) {
    if (tid < w) { rs[tid] += rs[tid + w]; rq[tid] += rq[tid + w]; }
    __syncthreads();
  }
  const float mu   = rs[0] * (1.0f / DMEM);
  const float var  = rq[0] * (1.0f / DMEM) - mu * mu;
  const float rstd = rsqrtf(var + 1e-5f);
  out[(size_t)row * DMEM + tid]       = (v0 - mu) * rstd * g[tid] + bb[tid];
  out[(size_t)row * DMEM + tid + 256] = (v1 - mu) * rstd * g[tid + 256] + bb[tid + 256];
}

// ---------------------------------------------------------------------------
extern "C" void kernel_launch(void* const* d_in, const int* in_sizes, int n_in,
                              void* d_out, int out_size, void* d_ws, size_t ws_size,
                              hipStream_t stream) {
  const float* adj       = (const float*)d_in[0];
  const float* input_emb = (const float*)d_in[1];
  // d_in[2] = seq_lens (all == N, unused)
  const float* W0_w = (const float*)d_in[3];
  const float* W0_b = (const float*)d_in[4];
  const float* W1_w = (const float*)d_in[5];
  const float* W1_b = (const float*)d_in[6];
  const float* ln_g = (const float*)d_in[7];
  const float* ln_b = (const float*)d_in[8];
  float* out = (float*)d_out;

  char* ws = (char*)d_ws;
  size_t off = 0;
  auto take = [&](size_t bytes) -> char* {
    char* p = ws + off;
    off += (bytes + 255) & ~(size_t)255;
    return p;
  };
  _Float16* adj_h = (_Float16*)take((size_t)B_ * N_ * N_ * 2);
  _Float16* x_h   = (_Float16*)take((size_t)B_ * N_ * DIN * 2);
  _Float16* xT_h  = (_Float16*)take((size_t)B_ * DIN * N_ * 2);
  _Float16* T_h   = (_Float16*)take((size_t)B_ * N_ * DIN * 2);
  _Float16* W0h   = (_Float16*)take((size_t)DMEM * DIN * 2);
  _Float16* W1h   = (_Float16*)take((size_t)DMEM * DMEM * 2);
  float*    denom = (float*)take((size_t)B_ * N_ * 4);
  float*    rowsm = (float*)take((size_t)B_ * N_ * 4);

  float* mask_out = out + (size_t)B_ * N_ * DMEM;

  // precompute: conversions, denom, masks
  k_adj_cvt_rowsum<<<dim3(B_ * N_), 256, 0, stream>>>(adj, adj_h, denom, rowsm);
  k_colsum_mask<<<dim3(N_ / 256, B_), 256, 0, stream>>>(adj, rowsm, mask_out);
  k_cvt_x_T<<<dim3(DIN / 64, N_ / 64, B_), 256, 0, stream>>>(input_emb, x_h, xT_h);
  k_f32_to_f16<<<(DMEM * DIN) / 256, 256, 0, stream>>>(W0_w, W0h, DMEM * DIN);
  k_f32_to_f16<<<(DMEM * DMEM) / 256, 256, 0, stream>>>(W1_w, W1h, DMEM * DMEM);

  // layer 0
  k_gemm1<<<dim3(DIN / 128, N_ / 128, B_), 256, 0, stream>>>(adj_h, xT_h, x_h, T_h);
  k_gemm2<<<dim3(DMEM / 128, (B_ * N_) / 128), 256, 0, stream>>>(T_h, W0h, W0_b, denom, x_h, xT_h);
  // layer 1
  k_gemm1<<<dim3(DMEM / 128, N_ / 128, B_), 256, 0, stream>>>(adj_h, xT_h, x_h, T_h);
  k_gemm2<<<dim3(DMEM / 128, (B_ * N_) / 128), 256, 0, stream>>>(T_h, W1h, W1_b, denom, x_h, xT_h);

  // final LayerNorm -> f32 output
  k_layernorm<<<dim3(B_ * N_), 256, 0, stream>>>(x_h, ln_g, ln_b, out);
}
